// DebertaV2DisentangledSelfAttention_13374528160410
// MI455X (gfx1250) — matrix operand order repaired and checked by reading, not compiled, not run on hardware
//
#include <hip/hip_runtime.h>
#include <hip/hip_bf16.h>

typedef __bf16 bf16;
typedef __attribute__((ext_vector_type(16))) __bf16 v16bf;
typedef __attribute__((ext_vector_type(8)))  float  v8f;

#define NH 16
#define HD 64
#define SEQ 1024
#define BATCH 4
#define HDIM 1024
#define RNUM 1025
#define MAXREL 512

static __device__ inline v8f wmma_bf16(v16bf a, v16bf b, v8f c) {
  return __builtin_amdgcn_wmma_f32_16x16x32_bf16(false, a, false, b, (short)0, c, false, false);
}

// A operand, 16x32 bf16 tile from row-major src. lane: M = lane&15, kh = lane>>4.
// element e -> K = kh*16+e
static __device__ inline v16bf load_a16(const bf16* base, int row0, int maxrow,
                                        int ld, int koff, int lane) {
  int m = lane & 15, kh = lane >> 4;
  int r = row0 + m; if (r > maxrow) r = maxrow;
  const bf16* p = base + (size_t)r * ld + koff + kh * 16;
  v16bf a;
#pragma unroll
  for (int e = 0; e < 8; ++e) { a[e] = p[e]; a[8 + e] = p[8 + e]; }
  return a;
}

// B operand, 32x16 bf16 tile. Source is "B-transposed": srcT row n holds column n
// contiguous in K. lane: N = lane&15, kh = lane>>4; element e -> K = kh*16+e.
static __device__ inline v16bf load_b16(const bf16* srcT, int nrow0, int ld,
                                        int koff, int lane) {
  int n = lane & 15, kh = lane >> 4;
  const bf16* p = srcT + (size_t)(nrow0 + n) * ld + koff + kh * 16;
  v16bf b;
#pragma unroll
  for (int e = 0; e < 16; ++e) b[e] = p[e];
  return b;
}

// B operand from rel table [R][NH*HD]: column n of the tile = rel row clip(dbase+n).
// Clamping here makes boundary buckets automatically correct for the diagonal gather.
static __device__ inline v16bf load_b_rel(const bf16* relHead, int dbase, int koff,
                                          int lane) {
  int n = lane & 15, kh = lane >> 4;
  int d = dbase + n;
  d = d < -MAXREL ? -MAXREL : (d > MAXREL ? MAXREL : d);
  const bf16* p = relHead + (size_t)(d + MAXREL) * (NH * HD) + koff + kh * 16;
  v16bf b;
#pragma unroll
  for (int e = 0; e < 16; ++e) b[e] = p[e];
  return b;
}

// ---------------- fp32 -> bf16 conversion (optionally transposed) ----------------
__global__ void cvt_bf16_kernel(const float* __restrict__ src, bf16* __restrict__ dst,
                                int rows, int cols, int transpose) {
  long i = (long)blockIdx.x * blockDim.x + threadIdx.x;
  long total = (long)rows * cols;
  if (i >= total) return;
  int r = (int)(i / cols), c = (int)(i % cols);
  float v = src[i];
  if (transpose) dst[(long)c * rows + r] = (bf16)v;
  else           dst[i] = (bf16)v;
}

// ---------------- bf16 WMMA GEMM:  out = X[M,1024] @ W[1024,1024] + bias ----------
// Wave tile 32(M) x 64(N). Software-pipelined: loads for K-step n+1 are issued
// before the 8 WMMAs of K-step n, so each iteration's wait covers loads issued a
// full iteration earlier (latency hidden inside a single wave, not just by occupancy).
// Wt is W transposed ([N][K]) so the B-operand load is contiguous.
// mode 0: out bf16 [B][NH][S][HD]   (Q,K)
// mode 1: out bf16 [B][NH][HD][S]   (V transposed)
// mode 2: out bf16 row-major [M][1024]  (rel_k / rel_q == [R][NH][HD])
__global__ __launch_bounds__(256) void gemm_bf16_kernel(
    const bf16* __restrict__ X, const bf16* __restrict__ Wt,
    const float* __restrict__ bias, bf16* __restrict__ out, int M, int mode) {
  int lane = threadIdx.x & 31;
  int w = blockIdx.x * 8 + (threadIdx.x >> 5);
  int tilesM = (M + 31) >> 5;
  if (w >= tilesM * 16) return;           // uniform per wave -> EXEC all ones inside
  int tm = w >> 4, tn = w & 15;
  int m0 = tm * 32, n0 = tn * 64;
  int maxrow = M - 1;

  v8f acc[2][4];
#pragma unroll
  for (int mt = 0; mt < 2; ++mt)
#pragma unroll
    for (int nt = 0; nt < 4; ++nt)
      acc[mt][nt] = (v8f){0,0,0,0,0,0,0,0};

  v16bf aC[2], bC[4], aN[2], bN[4];
#pragma unroll
  for (int mt = 0; mt < 2; ++mt)
    aC[mt] = load_a16(X, m0 + mt * 16, maxrow, HDIM, 0, lane);
#pragma unroll
  for (int nt = 0; nt < 4; ++nt)
    bC[nt] = load_b16(Wt, n0 + nt * 16, HDIM, 0, lane);

  for (int k0 = 0; k0 < HDIM; k0 += 32) {
    int kn = (k0 + 32 < HDIM) ? (k0 + 32) : 0;   // last-iter prefetch is dummy
#pragma unroll
    for (int mt = 0; mt < 2; ++mt)
      aN[mt] = load_a16(X, m0 + mt * 16, maxrow, HDIM, kn, lane);
#pragma unroll
    for (int nt = 0; nt < 4; ++nt)
      bN[nt] = load_b16(Wt, n0 + nt * 16, HDIM, kn, lane);

#pragma unroll
    for (int mt = 0; mt < 2; ++mt)
#pragma unroll
      for (int nt = 0; nt < 4; ++nt)
        acc[mt][nt] = wmma_bf16(aC[mt], bC[nt], acc[mt][nt]);

#pragma unroll
    for (int mt = 0; mt < 2; ++mt) aC[mt] = aN[mt];
#pragma unroll
    for (int nt = 0; nt < 4; ++nt) bC[nt] = bN[nt];
  }

  int half = lane >> 4, nl = lane & 15;
#pragma unroll
  for (int nt = 0; nt < 4; ++nt) {
    int gn = n0 + nt * 16 + nl;
    float bb = bias[gn];
#pragma unroll
    for (int mt = 0; mt < 2; ++mt) {
#pragma unroll
      for (int r = 0; r < 8; ++r) {
        int m = m0 + mt * 16 + r + 8 * half;
        if (m >= M) continue;
        float v = acc[mt][nt][r] + bb;
        size_t idx;
        if (mode == 0) {
          int b4 = m >> 10, s = m & 1023, head = gn >> 6, h = gn & 63;
          idx = (((size_t)(b4 * NH + head) * SEQ) + s) * HD + h;
        } else if (mode == 1) {
          int b4 = m >> 10, s = m & 1023, head = gn >> 6, h = gn & 63;
          idx = (((size_t)(b4 * NH + head) * HD) + h) * SEQ + s;
        } else {
          idx = (size_t)m * HDIM + gn;
        }
        out[idx] = (bf16)v;
      }
    }
  }
}

// ---------------- fused disentangled flash attention ----------------
// grid: (S/64, NH, B), block: 128 (4 waves). Each wave owns one 16-row i-tile.
__global__ __launch_bounds__(128) void deberta_attn_kernel(
    const bf16* __restrict__ Q, const bf16* __restrict__ K,
    const bf16* __restrict__ Vt, const bf16* __restrict__ RK,
    const bf16* __restrict__ RQ, const float* __restrict__ mask,
    float* __restrict__ out) {
  constexpr float SCALE = 0.07216878364870323f;   // 1/sqrt(3*64)

  __shared__ float stA[4][16][48];   // c2p staging  (per-wave private)
  __shared__ float stB[4][32][32];   // p2c staging
  __shared__ bf16  stP[4][16][32];   // P f32(Clayout) -> bf16 A-layout transpose

  int lane = threadIdx.x & 31;
  int wave = threadIdx.x >> 5;
  int half = lane >> 4, nl = lane & 15;
  int i0 = blockIdx.x * 64 + wave * 16;
  int head = blockIdx.y, b = blockIdx.z;

  const bf16* Qb  = Q  + ((size_t)(b * NH + head)) * SEQ * HD;
  const bf16* Kb  = K  + ((size_t)(b * NH + head)) * SEQ * HD;
  const bf16* Vb  = Vt + ((size_t)(b * NH + head)) * HD * SEQ;
  const bf16* RKh = RK + head * HD;
  const bf16* RQh = RQ + head * HD;
  const float* mk = mask + (size_t)b * SEQ;

  v16bf aq0 = load_a16(Qb, i0, SEQ - 1, HD, 0,  lane);
  v16bf aq1 = load_a16(Qb, i0, SEQ - 1, HD, 32, lane);

  v8f o[4] = {{0,0,0,0,0,0,0,0},{0,0,0,0,0,0,0,0},
              {0,0,0,0,0,0,0,0},{0,0,0,0,0,0,0,0}};
  float mrow[8], lrow[8];
#pragma unroll
  for (int r = 0; r < 8; ++r) { mrow[r] = -1.0e30f; lrow[r] = 0.0f; }

  for (int j0 = 0; j0 < SEQ; j0 += 32) {
    // ---- c2c : Q(16x64) @ K^T for two 16-col j sub-tiles ----
    v8f s0 = {0,0,0,0,0,0,0,0}, s1 = {0,0,0,0,0,0,0,0};
    {
      v16bf bk0 = load_b16(Kb, j0,      HD, 0,  lane);
      v16bf bk1 = load_b16(Kb, j0,      HD, 32, lane);
      s0 = wmma_bf16(aq0, bk0, s0); s0 = wmma_bf16(aq1, bk1, s0);
      v16bf bk2 = load_b16(Kb, j0 + 16, HD, 0,  lane);
      v16bf bk3 = load_b16(Kb, j0 + 16, HD, 32, lane);
      s1 = wmma_bf16(aq0, bk2, s1); s1 = wmma_bf16(aq1, bk3, s1);
    }

    // ---- c2p : Q @ relK^T over 48 consecutive (clamped) buckets ----
    int dbase = i0 - j0 - 31;
#pragma unroll
    for (int nt = 0; nt < 3; ++nt) {
      v8f pc = {0,0,0,0,0,0,0,0};
      pc = wmma_bf16(aq0, load_b_rel(RKh, dbase + nt * 16, 0,  lane), pc);
      pc = wmma_bf16(aq1, load_b_rel(RKh, dbase + nt * 16, 32, lane), pc);
#pragma unroll
      for (int r = 0; r < 8; ++r)
        stA[wave][r + 8 * half][nt * 16 + nl] = pc[r];
    }

    // ---- p2c : K @ relQ^T, 31 (clamped) buckets per j sub-tile ----
#pragma unroll
    for (int jt = 0; jt < 2; ++jt) {
      v16bf ak0 = load_a16(Kb, j0 + jt * 16, SEQ - 1, HD, 0,  lane);
      v16bf ak1 = load_a16(Kb, j0 + jt * 16, SEQ - 1, HD, 32, lane);
      int dbase2 = (j0 + jt * 16) - i0 - 15;
#pragma unroll
      for (int nt = 0; nt < 2; ++nt) {
        v8f pp = {0,0,0,0,0,0,0,0};
        pp = wmma_bf16(ak0, load_b_rel(RQh, dbase2 + nt * 16, 0,  lane), pp);
        pp = wmma_bf16(ak1, load_b_rel(RQh, dbase2 + nt * 16, 32, lane), pp);
#pragma unroll
        for (int r = 0; r < 8; ++r)
          stB[wave][jt * 16 + r + 8 * half][nt * 16 + nl] = pp[r];
      }
    }
    __syncthreads();

    // ---- diagonal gather + scale + mask ----
    float mj0 = mk[j0 + nl], mj1 = mk[j0 + 16 + nl];
#pragma unroll
    for (int r = 0; r < 8; ++r) {
      int mi = r + 8 * half;
      // jt=0: jj = nl      -> c2p col = mi-nl+31, p2c row jj, col = nl-mi+15
      s0[r] = (s0[r] + stA[wave][mi][mi - nl + 31]
                     + stB[wave][nl][nl - mi + 15]) * SCALE + mj0;
      // jt=1: jj = 16+nl   -> c2p col = mi-nl+15, p2c row 16+nl, col = nl-mi+15
      s1[r] = (s1[r] + stA[wave][mi][mi - nl + 15]
                     + stB[wave][16 + nl][nl - mi + 15]) * SCALE + mj1;
    }

    // ---- online softmax (row stats replicated across 16-lane half) ----
#pragma unroll
    for (int r = 0; r < 8; ++r) {
      float t = fmaxf(s0[r], s1[r]);
      t = fmaxf(t, __shfl_xor(t, 1, 32));
      t = fmaxf(t, __shfl_xor(t, 2, 32));
      t = fmaxf(t, __shfl_xor(t, 4, 32));
      t = fmaxf(t, __shfl_xor(t, 8, 32));
      float mnew = fmaxf(mrow[r], t);
      float alpha = __expf(mrow[r] - mnew);
      mrow[r] = mnew;
      s0[r] = __expf(s0[r] - mnew);
      s1[r] = __expf(s1[r] - mnew);
      float rs = s0[r] + s1[r];
      rs += __shfl_xor(rs, 1, 32);
      rs += __shfl_xor(rs, 2, 32);
      rs += __shfl_xor(rs, 4, 32);
      rs += __shfl_xor(rs, 8, 32);
      lrow[r] = lrow[r] * alpha + rs;
#pragma unroll
      for (int ht = 0; ht < 4; ++ht) o[ht][r] *= alpha;
      int mi = r + 8 * half;
      stP[wave][mi][nl]      = (bf16)s0[r];
      stP[wave][mi][16 + nl] = (bf16)s1[r];
    }
    __syncthreads();

    // ---- P (16x32, A-layout from LDS) @ V^T ----
    v16bf ap;
    {
      int m = nl, kh = half;
#pragma unroll
      for (int e = 0; e < 8; ++e) {
        ap[e]     = stP[wave][m][kh * 16 + e];
        ap[8 + e] = stP[wave][m][kh * 16 + 8 + e];
      }
    }
#pragma unroll
    for (int ht = 0; ht < 4; ++ht) {
      v16bf bv = load_b16(Vb, ht * 16, SEQ, j0, lane);
      o[ht] = wmma_bf16(ap, bv, o[ht]);
    }
  }

  // ---- normalize + store ctx [B,S,H] fp32 ----
#pragma unroll
  for (int r = 0; r < 8; ++r) {
    float inv = 1.0f / lrow[r];
    int s = i0 + r + 8 * half;
    float* op = out + ((size_t)(b * SEQ + s)) * HDIM + head * HD;
#pragma unroll
    for (int ht = 0; ht < 4; ++ht)
      op[ht * 16 + nl] = o[ht][r] * inv;
  }
}

// ---------------- host ----------------
extern "C" void kernel_launch(void* const* d_in, const int* in_sizes, int n_in,
                              void* d_out, int out_size, void* d_ws, size_t ws_size,
                              hipStream_t stream) {
  (void)in_sizes; (void)n_in; (void)out_size; (void)ws_size;
  const float* hidden = (const float*)d_in[0];
  const float* amask  = (const float*)d_in[1];
  const float* rel    = (const float*)d_in[2];
  const float* Wq  = (const float*)d_in[3];  const float* bq  = (const float*)d_in[4];
  const float* Wk  = (const float*)d_in[5];  const float* bk  = (const float*)d_in[6];
  const float* Wv  = (const float*)d_in[7];  const float* bv  = (const float*)d_in[8];
  const float* Wpk = (const float*)d_in[9];  const float* bpk = (const float*)d_in[10];
  const float* Wpq = (const float*)d_in[11]; const float* bpq = (const float*)d_in[12];
  float* out = (float*)d_out;

  char* ws = (char*)d_ws;
  size_t off = 0;
  auto alloc = [&](size_t bytes) -> void* {
    void* p = ws + off; off += (bytes + 255) & ~(size_t)255; return p;
  };
  const size_t ROWS = (size_t)BATCH * SEQ;               // 4096
  bf16* Xb   = (bf16*)alloc(ROWS * HDIM * 2);
  bf16* Rb   = (bf16*)alloc((size_t)RNUM * HDIM * 2);
  bf16* WtQ  = (bf16*)alloc((size_t)HDIM * HDIM * 2);
  bf16* WtK  = (bf16*)alloc((size_t)HDIM * HDIM * 2);
  bf16* WtV  = (bf16*)alloc((size_t)HDIM * HDIM * 2);
  bf16* WtPK = (bf16*)alloc((size_t)HDIM * HDIM * 2);
  bf16* WtPQ = (bf16*)alloc((size_t)HDIM * HDIM * 2);
  bf16* Qb   = (bf16*)alloc(ROWS * HDIM * 2);
  bf16* Kb   = (bf16*)alloc(ROWS * HDIM * 2);
  bf16* Vtb  = (bf16*)alloc(ROWS * HDIM * 2);
  bf16* RKb  = (bf16*)alloc((size_t)RNUM * HDIM * 2);
  bf16* RQb  = (bf16*)alloc((size_t)RNUM * HDIM * 2);

  auto cvt = [&](const float* s, bf16* d, int rows, int cols, int tr) {
    long total = (long)rows * cols;
    cvt_bf16_kernel<<<(unsigned)((total + 255) / 256), 256, 0, stream>>>(s, d, rows, cols, tr);
  };
  cvt(hidden, Xb, (int)ROWS, HDIM, 0);
  cvt(rel,    Rb, RNUM,      HDIM, 0);
  cvt(Wq,  WtQ,  HDIM, HDIM, 1);
  cvt(Wk,  WtK,  HDIM, HDIM, 1);
  cvt(Wv,  WtV,  HDIM, HDIM, 1);
  cvt(Wpk, WtPK, HDIM, HDIM, 1);
  cvt(Wpq, WtPQ, HDIM, HDIM, 1);

  auto gemm = [&](const bf16* X, const bf16* Wt, const float* bias, bf16* o,
                  int M, int mode) {
    int tilesM = (M + 31) >> 5;
    int waves  = tilesM * 16;                 // N/64 = 16 n-tiles
    int blocks = (waves + 7) / 8;
    gemm_bf16_kernel<<<blocks, 256, 0, stream>>>(X, Wt, bias, o, M, mode);
  };
  gemm(Xb, WtQ,  bq,  Qb,  (int)ROWS, 0);
  gemm(Xb, WtK,  bk,  Kb,  (int)ROWS, 0);
  gemm(Xb, WtV,  bv,  Vtb, (int)ROWS, 1);
  gemm(Rb, WtPK, bpk, RKb, RNUM, 2);
  gemm(Rb, WtPQ, bpq, RQb, RNUM, 2);

  dim3 grid(SEQ / 64, NH, BATCH);
  deberta_attn_kernel<<<grid, 128, 0, stream>>>(Qb, Kb, Vtb, RKb, RQb, amask, out);
}
